// GCSMixer_17239998726335
// MI455X (gfx1250) — matrix operand order repaired and checked by reading, not compiled, not run on hardware
//
#include <hip/hip_runtime.h>
#include <math.h>

// ---------------------------------------------------------------------------
// GCSMixer for MI455X (gfx1250, wave32, WMMA).
// bf16 WMMA (f32 accumulate) for all channel-mixing linears; per-channel
// spatial shifts, GELU/ReLU, residuals, LayerNorm and BatchNorm-affine fused
// into GEMM prologues/epilogues. Fragment loads vectorized (ds_load_b128);
// elementwise kernels vectorized 4-wide across contiguous channels.
// ---------------------------------------------------------------------------

typedef __attribute__((ext_vector_type(4)))  __bf16 v4bf;
typedef __attribute__((ext_vector_type(8)))  __bf16 v8bf;
typedef __attribute__((ext_vector_type(16))) __bf16 v16bf;
typedef __attribute__((ext_vector_type(4)))  float  v4f;
typedef __attribute__((ext_vector_type(8)))  float  v8f;

#define CC    96
#define NN    32768          // 32*32*32
#define NTOK  262144         // B * N  (B = 8)

__device__ __forceinline__ float gelu_exact(float v) {
    return 0.5f * v * (1.0f + erff(v * 0.70710678118654752440f));
}

// shift: s[c] = (c/32 + c%32) % 32   (identical for D/H/W since g==32)
__device__ __forceinline__ int shift_of(int c) {
    return ((c >> 5) + (c & 31)) & 31;
}

// Per-lane WMMA tile compute over KTILES k-steps and 6 n-tiles.
// A: two contiguous 16B runs per lane; B: contiguous 32B per lane (pre-swizzled).
#define WMMA_TILE_COMPUTE(ATILE, ASTR, BSWZ, KTILES)                           \
    {                                                                          \
        _Pragma("unroll")                                                      \
        for (int kt = 0; kt < (KTILES); ++kt) {                                \
            const int abase = (wrow + lanelow) * (ASTR) + kt * 32 + kgrp * 8;  \
            const v8bf alo = *(const v8bf*)&(ATILE)[abase];                    \
            const v8bf ahi = *(const v8bf*)&(ATILE)[abase + 16];               \
            const v16bf a = __builtin_shufflevector(                           \
                alo, ahi, 0, 1, 2, 3, 4, 5, 6, 7,                              \
                8, 9, 10, 11, 12, 13, 14, 15);                                 \
            _Pragma("unroll")                                                  \
            for (int nt = 0; nt < 6; ++nt) {                                   \
                const v16bf bm =                                               \
                    *(const v16bf*)&(BSWZ)[((kt * 6 + nt) * 32 + lane) * 16];  \
                acc[nt] = __builtin_amdgcn_wmma_f32_16x16x32_bf16(             \
                    false, a, false, bm, (short)0, acc[nt], false, false);     \
            }                                                                  \
        }                                                                      \
    }

// Swizzled weight load: bswz[((kt*6+nt)*32+lane)*16 + j] = w[n][k],
// k = kt*32 + (j/8)*16 + (lane/16)*8 + j%8, n = nt*16 + (lane%16).
#define LOAD_BSWZ(BSWZ, WPTR, KTILES, KDIM, NTHREADS)                          \
    for (int i = tid; i < (KTILES) * 6 * 32 * 16; i += (NTHREADS)) {           \
        const int j   = i & 15;                                                \
        const int ln  = (i >> 4) & 31;                                         \
        const int ntk = i >> 9;                                                \
        const int kt  = ntk / 6, nt = ntk - kt * 6;                            \
        const int k   = kt * 32 + (j & 8) * 2 + (ln >> 4) * 8 + (j & 7);      \
        const int n   = nt * 16 + (ln & 15);                                   \
        (BSWZ)[i] = (__bf16)(WPTR)[n * (KDIM) + k];                            \
    }

// ---------------------------------------------------------------------------
// Generic fused shift-gather + GEMM (out = act(gather(in) @ w.T + b) [+res]).
// AXIS: 0=D, 1=H, 2=W, 3=no gather.  ACT: 0=none, 1=gelu, 2=relu.
// Tile: 128 tokens x 96 outputs per block; 8 waves, each 16 rows x 6 n-tiles.
// ---------------------------------------------------------------------------
template<int AXIS, int DIR, int ACT, bool RESID, typename InT, typename OutT>
__global__ void __launch_bounds__(256)
shift_gemm(const InT* __restrict__ in, const float* __restrict__ w,
           const float* __restrict__ bias, const float* __restrict__ resid,
           OutT* __restrict__ out)
{
    __shared__ __attribute__((aligned(32))) __bf16 bswz[3 * 6 * 32 * 16];
    __shared__ __attribute__((aligned(32))) __bf16 atile[128 * 104];

    const int tid  = threadIdx.x;
    const int row0 = blockIdx.x * 128;

    LOAD_BSWZ(bswz, w, 3, 96, 256)

    // gathered A tile -> LDS (bf16)
    if (AXIS == 3 && sizeof(InT) == 2) {
        // identity gather, bf16 input: vectorized 4-channel copy
        for (int idx = tid; idx < 128 * 24; idx += 256) {
            const int r = idx / 24, c0 = (idx - r * 24) * 4;
            const int t = row0 + r;
            const v4bf v = *(const v4bf*)((const __bf16*)in + (size_t)t * CC + c0);
            *(v4bf*)&atile[r * 104 + c0] = v;
        }
    } else {
        for (int idx = tid; idx < 128 * 96; idx += 256) {
            const int r = idx / 96, c = idx - r * 96;
            const int t = row0 + r;
            const int b = t >> 15;
            const int n = t & (NN - 1);
            int sd = n >> 10, sh = (n >> 5) & 31, sw = n & 31;
            if (AXIS == 0) sd = (sd - DIR * shift_of(c)) & 31;
            if (AXIS == 1) sh = (sh - DIR * shift_of(c)) & 31;
            if (AXIS == 2) sw = (sw - DIR * shift_of(c)) & 31;
            const int nsrc = (sd << 10) | (sh << 5) | sw;
            const size_t src = ((size_t)((b << 15) | nsrc)) * CC + c;
            atile[r * 104 + c] = (__bf16)((float)in[src]);
        }
    }
    __syncthreads();

    const int wid = tid >> 5, lane = tid & 31;
    const int lanelow = lane & 15, kgrp = lane >> 4;
    const int wrow = wid * 16;

    v8f acc[6];
    #pragma unroll
    for (int i = 0; i < 6; ++i)
        #pragma unroll
        for (int r = 0; r < 8; ++r) acc[i][r] = 0.0f;

    WMMA_TILE_COMPUTE(atile, 104, bswz, 3)

    #pragma unroll
    for (int nt = 0; nt < 6; ++nt) {
        const int ncol = nt * 16 + lanelow;
        const float bv = bias[ncol];
        #pragma unroll
        for (int r = 0; r < 8; ++r) {
            const int token = row0 + wrow + r + 8 * kgrp;
            float v = acc[nt][r] + bv;
            if (ACT == 1) v = gelu_exact(v);
            else if (ACT == 2) v = fmaxf(v, 0.0f);
            if (RESID) v += resid[(size_t)token * CC + ncol];
            out[(size_t)token * CC + ncol] = (OutT)v;
        }
    }
}

// ---------------------------------------------------------------------------
// LayerNorm(concat(x1,x2)) @ wf.T + bf + x   (K = 192, 64 tokens / block)
// ---------------------------------------------------------------------------
__global__ void __launch_bounds__(128)
lnwf_gemm(const float* __restrict__ x1, const float* __restrict__ x2,
          const float* __restrict__ lng, const float* __restrict__ lnb,
          const float* __restrict__ wf, const float* __restrict__ bfv,
          const float* __restrict__ xres, float* __restrict__ xf)
{
    __shared__ __attribute__((aligned(32))) __bf16 alds[64 * 200];
    __shared__ __attribute__((aligned(32))) __bf16 bswz[6 * 6 * 32 * 16];

    const int tid  = threadIdx.x;
    const int row0 = blockIdx.x * 64;

    LOAD_BSWZ(bswz, wf, 6, 192, 128)

    if (tid < 64) {
        const int token = row0 + tid;
        const float* p1 = x1 + (size_t)token * CC;
        const float* p2 = x2 + (size_t)token * CC;
        float s = 0.0f, sq = 0.0f;
        #pragma unroll
        for (int c = 0; c < 96; ++c) { const float v = p1[c]; s += v; sq += v * v; }
        #pragma unroll
        for (int c = 0; c < 96; ++c) { const float v = p2[c]; s += v; sq += v * v; }
        const float mean = s * (1.0f / 192.0f);
        const float var  = sq * (1.0f / 192.0f) - mean * mean;
        const float inv  = rsqrtf(var + 1e-6f);
        #pragma unroll
        for (int c = 0; c < 96; ++c) {
            const float nv = (p1[c] - mean) * inv * lng[c] + lnb[c];
            alds[tid * 200 + c] = (__bf16)nv;
        }
        #pragma unroll
        for (int c = 0; c < 96; ++c) {
            const float nv = (p2[c] - mean) * inv * lng[96 + c] + lnb[96 + c];
            alds[tid * 200 + 96 + c] = (__bf16)nv;
        }
    }
    __syncthreads();

    const int wid = tid >> 5, lane = tid & 31;
    const int lanelow = lane & 15, kgrp = lane >> 4;
    const int wrow = wid * 16;

    v8f acc[6];
    #pragma unroll
    for (int i = 0; i < 6; ++i)
        #pragma unroll
        for (int r = 0; r < 8; ++r) acc[i][r] = 0.0f;

    WMMA_TILE_COMPUTE(alds, 200, bswz, 6)

    #pragma unroll
    for (int nt = 0; nt < 6; ++nt) {
        const int ncol = nt * 16 + lanelow;
        const float bv = bfv[ncol];
        #pragma unroll
        for (int r = 0; r < 8; ++r) {
            const int token = row0 + wrow + r + 8 * kgrp;
            xf[(size_t)token * CC + ncol] =
                acc[nt][r] + bv + xres[(size_t)token * CC + ncol];
        }
    }
}

// ---------------------------------------------------------------------------
// depthwise 3x3x3 conv (zero pad), bf16 output.
// One thread per (token, 4-channel group): neighbor addresses are
// channel-independent, loads are 16B vectors, store is one 8B vector.
// ---------------------------------------------------------------------------
__global__ void __launch_bounds__(256)
dwconv(const float* __restrict__ x, const float* __restrict__ kq,
       const float* __restrict__ kb, __bf16* __restrict__ xdw)
{
    __shared__ __attribute__((aligned(16))) float skt[27 * 96];  // [tap][c]
    __shared__ __attribute__((aligned(16))) float sb[96];
    const int tid = threadIdx.x;
    for (int i = tid; i < 27 * 96; i += 256) {
        const int tap = i / 96, c = i - tap * 96;
        skt[i] = kq[c * 27 + tap];
    }
    if (tid < 96) sb[tid] = kb[tid];
    __syncthreads();

    const int gid = blockIdx.x * 256 + tid;           // NTOK*24 threads
    const int c0 = (gid % 24) * 4;
    const int t = gid / 24;
    const int b = t >> 15;
    const int n = t & (NN - 1);
    const int d = n >> 10, h = (n >> 5) & 31, w = n & 31;

    v4f acc4 = *(const v4f*)&sb[c0];
    #pragma unroll
    for (int di = -1; di <= 1; ++di) {
        const int dd = d + di;
        const bool okd = (unsigned)dd < 32u;
        #pragma unroll
        for (int hj = -1; hj <= 1; ++hj) {
            const int hh = h + hj;
            const bool okh = (unsigned)hh < 32u;
            #pragma unroll
            for (int wk = -1; wk <= 1; ++wk) {
                const int ww = w + wk;
                if (okd && okh && (unsigned)ww < 32u) {
                    const size_t src =
                        ((size_t)((b << 15) | (dd << 10) | (hh << 5) | ww)) * CC + c0;
                    const int tap = (di + 1) * 9 + (hj + 1) * 3 + (wk + 1);
                    const v4f xv = *(const v4f*)&x[src];
                    const v4f kv = *(const v4f*)&skt[tap * 96 + c0];
                    acc4 += xv * kv;
                }
            }
        }
    }
    v4bf o;
    #pragma unroll
    for (int i = 0; i < 4; ++i) o[i] = (__bf16)acc4[i];
    *(v4bf*)&xdw[(size_t)t * CC + c0] = o;
}

// ---------------------------------------------------------------------------
// BatchNorm statistics (sum / sumsq per channel), atomically accumulated
// ---------------------------------------------------------------------------
__global__ void bn_zero(float* stats) {
    const int i = blockIdx.x * blockDim.x + threadIdx.x;
    if (i < 192) stats[i] = 0.0f;
}

__global__ void __launch_bounds__(32)
bn_stats(const float* __restrict__ xl, float* __restrict__ stats)
{
    const int t0 = blockIdx.x * 256;
    const int q = threadIdx.x;
    if (q >= 24) return;
    const int c0 = q * 4;
    v4f s = {0.0f, 0.0f, 0.0f, 0.0f};
    v4f sq = {0.0f, 0.0f, 0.0f, 0.0f};
    for (int i = 0; i < 256; ++i) {
        const v4f v = *(const v4f*)&xl[(size_t)(t0 + i) * CC + c0];
        s += v; sq += v * v;
    }
    #pragma unroll
    for (int k = 0; k < 4; ++k) {
        atomicAdd(&stats[c0 + k], s[k]);
        atomicAdd(&stats[96 + c0 + k], sq[k]);
    }
}

// ---------------------------------------------------------------------------
// out = concat(xf, BN(xl)) @ wo.T + bo   (K = 192, BN-affine fused in A build)
// ---------------------------------------------------------------------------
__global__ void __launch_bounds__(128)
wo_gemm(const float* __restrict__ xf, const float* __restrict__ xl,
        const float* __restrict__ stats, const float* __restrict__ bng,
        const float* __restrict__ bnb, const float* __restrict__ wo,
        const float* __restrict__ bo, float* __restrict__ out)
{
    __shared__ __attribute__((aligned(32))) __bf16 alds[64 * 200];
    __shared__ __attribute__((aligned(32))) __bf16 bswz[6 * 6 * 32 * 16];
    __shared__ float bscale[96];
    __shared__ float bshift[96];

    const int tid  = threadIdx.x;
    const int row0 = blockIdx.x * 64;

    if (tid < 96) {
        const float cnt = (float)NTOK;
        const float m   = stats[tid] / cnt;
        const float v   = stats[96 + tid] / cnt - m * m;
        const float inv = rsqrtf(v + 1e-5f);
        bscale[tid] = inv * bng[tid];
        bshift[tid] = bnb[tid] - m * inv * bng[tid];
    }
    LOAD_BSWZ(bswz, wo, 6, 192, 128)
    __syncthreads();

    if (tid < 64) {
        const int token = row0 + tid;
        const float* pf = xf + (size_t)token * CC;
        const float* pl = xl + (size_t)token * CC;
        #pragma unroll
        for (int c = 0; c < 96; ++c)
            alds[tid * 200 + c] = (__bf16)pf[c];
        #pragma unroll
        for (int c = 0; c < 96; ++c)
            alds[tid * 200 + 96 + c] = (__bf16)(pl[c] * bscale[c] + bshift[c]);
    }
    __syncthreads();

    const int wid = tid >> 5, lane = tid & 31;
    const int lanelow = lane & 15, kgrp = lane >> 4;
    const int wrow = wid * 16;

    v8f acc[6];
    #pragma unroll
    for (int i = 0; i < 6; ++i)
        #pragma unroll
        for (int r = 0; r < 8; ++r) acc[i][r] = 0.0f;

    WMMA_TILE_COMPUTE(alds, 200, bswz, 6)

    #pragma unroll
    for (int nt = 0; nt < 6; ++nt) {
        const int ncol = nt * 16 + lanelow;
        const float bv = bo[ncol];
        #pragma unroll
        for (int r = 0; r < 8; ++r) {
            const int token = row0 + wrow + r + 8 * kgrp;
            out[(size_t)token * CC + ncol] = acc[nt][r] + bv;
        }
    }
}

// ---------------------------------------------------------------------------
extern "C" void kernel_launch(void* const* d_in, const int* in_sizes, int n_in,
                              void* d_out, int out_size, void* d_ws, size_t ws_size,
                              hipStream_t stream)
{
    (void)in_sizes; (void)n_in; (void)out_size; (void)ws_size;

    const float* x    = (const float*)d_in[0];
    const float* w1d  = (const float*)d_in[4];
    const float* b1d  = (const float*)d_in[5];
    const float* w1h  = (const float*)d_in[6];
    const float* b1h  = (const float*)d_in[7];
    const float* w1w  = (const float*)d_in[8];
    const float* b1w  = (const float*)d_in[9];
    const float* w2w  = (const float*)d_in[10];
    const float* b2w  = (const float*)d_in[11];
    const float* w2h  = (const float*)d_in[12];
    const float* b2h  = (const float*)d_in[13];
    const float* w2d  = (const float*)d_in[14];
    const float* b2d  = (const float*)d_in[15];
    const float* ln_g = (const float*)d_in[16];
    const float* ln_b = (const float*)d_in[17];
    const float* wf   = (const float*)d_in[18];
    const float* bfv  = (const float*)d_in[19];
    const float* wo   = (const float*)d_in[20];
    const float* bo   = (const float*)d_in[21];
    const float* dw_k = (const float*)d_in[22];
    const float* dw_b = (const float*)d_in[23];
    const float* pw_k = (const float*)d_in[24];
    const float* pw_b = (const float*)d_in[25];
    const float* bn_g = (const float*)d_in[26];
    const float* bn_b = (const float*)d_in[27];

    char* ws = (char*)d_ws;
    const size_t BF = (size_t)NTOK * CC * sizeof(__bf16);   //  48 MB
    const size_t FL = (size_t)NTOK * CC * sizeof(float);    //  96 MB
    __bf16* t1 = (__bf16*)(ws);
    __bf16* t2 = (__bf16*)(ws + BF);
    float*  x1 = (float*)(ws + 2 * BF);
    float*  x2 = (float*)(ws + 2 * BF + FL);
    float*  xf = (float*)(ws + 2 * BF + 2 * FL);
    float*  xl = x1;                   // x1 dead after lnwf_gemm
    __bf16* xdw = t1;                  // t1 dead after stage 5
    float*  stats = (float*)(ws + 2 * BF + 3 * FL);
    float*  out = (float*)d_out;

    // path 1: D -> H -> W, forward shifts
    shift_gemm<0, +1, 1, false, float,  __bf16><<<2048, 256, 0, stream>>>(x,  w1d, b1d, nullptr, t1);
    shift_gemm<1, +1, 1, false, __bf16, __bf16><<<2048, 256, 0, stream>>>(t1, w1h, b1h, nullptr, t2);
    shift_gemm<2, +1, 0, true,  __bf16, float ><<<2048, 256, 0, stream>>>(t2, w1w, b1w, x,       x1);
    // path 2: W -> H -> D, reverse shifts
    shift_gemm<2, -1, 1, false, float,  __bf16><<<2048, 256, 0, stream>>>(x,  w2w, b2w, nullptr, t1);
    shift_gemm<1, -1, 1, false, __bf16, __bf16><<<2048, 256, 0, stream>>>(t1, w2h, b2h, nullptr, t2);
    shift_gemm<0, -1, 0, true,  __bf16, float ><<<2048, 256, 0, stream>>>(t2, w2d, b2d, x,       x2);
    // LayerNorm + wf + residual
    lnwf_gemm<<<4096, 128, 0, stream>>>(x1, x2, ln_g, ln_b, wf, bfv, x, xf);
    // local branch: depthwise conv -> pointwise (WMMA) + ReLU -> BN stats
    dwconv<<<24576, 256, 0, stream>>>(x, dw_k, dw_b, xdw);
    shift_gemm<3, 0, 2, false, __bf16, float><<<2048, 256, 0, stream>>>(xdw, pw_k, pw_b, nullptr, xl);
    bn_zero<<<1, 192, 0, stream>>>(stats);
    bn_stats<<<1024, 32, 0, stream>>>(xl, stats);
    // final: concat(xf, BN(xl)) @ wo.T + bo
    wo_gemm<<<4096, 128, 0, stream>>>(xf, xl, stats, bn_g, bn_b, wo, bo, out);
}
